// GraphModel_37666863186542
// MI455X (gfx1250) — matrix operand order repaired and checked by reading, not compiled, not run on hardware
//
#include <hip/hip_runtime.h>

typedef __attribute__((ext_vector_type(16))) _Float16 v16h;
typedef __attribute__((ext_vector_type(8)))  float    v8f;

#define NNODES 50000
#define NEDGES 800000
#define C  64
#define C2 128
#define LDP 136   // padded LDS row stride (halves): 272B rows -> conflict-free b128

// workspace layout (bytes)
#define WLIN_OFF  ((size_t)0)                         // 128*128 f16 = 32 KB
#define WOUT_OFF  ((size_t)32768)                     // 64*128  f16 = 16 KB
#define NMAX_OFF  ((size_t)65536)                     // N*128 u32/f32 = 25.6 MB
#define DEN_OFF   (NMAX_OFF + (size_t)NNODES * C2 * 4)
#define NUM_OFF   (DEN_OFF  + (size_t)NNODES * C2 * 4)

union Frag { v16h v; uint4 q[2]; };

// ---------------- K0: zero accumulators, convert weights to f16 ----------------
__global__ __launch_bounds__(256) void init_kernel(
    const float* __restrict__ Wlin, const float* __restrict__ Wout,
    unsigned short* __restrict__ wlin_h, unsigned short* __restrict__ wout_h,
    unsigned* __restrict__ zero_base, long zero_dwords) {
  long i = (long)blockIdx.x * blockDim.x + threadIdx.x;
  long stride = (long)gridDim.x * blockDim.x;
  for (long j = i; j < zero_dwords; j += stride) zero_base[j] = 0u;
  if (i < C2 * C2) {
    _Float16 h = (_Float16)Wlin[i];
    __builtin_memcpy(&wlin_h[i], &h, 2);
  }
  if (i < C * C2) {
    _Float16 h = (_Float16)Wout[i];
    __builtin_memcpy(&wout_h[i], &h, 2);
  }
}

// ---------------- K1/K2: edge GEMM (att = relu(ef @ Wlin^T + b)) ----------------
// PASS 1: atomic u32 max of att bits (att >= 0 so uint order == float order)
// PASS 2: recompute att, ex = exp(att - mx[dst]); atomicAdd den, num
template <int PASS>
__global__ __launch_bounds__(256) void edge_kernel(
    const float* __restrict__ features,
    const unsigned short* __restrict__ wlin_u,
    const float* __restrict__ b_lin,
    const int* __restrict__ src, const int* __restrict__ dst,
    unsigned* __restrict__ nmax_u, const float* __restrict__ nmax_f,
    float* __restrict__ den, float* __restrict__ num) {
  __shared__ __align__(16) _Float16 ef[128 * LDP];
  __shared__ int s_src[128], s_dst[128];
  const _Float16* wlin = (const _Float16*)wlin_u;

  const int tid = threadIdx.x;
  const int ebase = blockIdx.x * 128;

  if (tid < 128) {
    s_src[tid] = src[ebase + tid];
    s_dst[tid] = dst[ebase + tid];
  }
  __syncthreads();

  { // gather ef = concat(f[src], f[dst]) -> LDS f16. thread = (row, half)
    const int r = tid >> 1, h = tid & 1;
    const int node = h ? s_dst[r] : s_src[r];
    const float4* fp = (const float4*)(features + (size_t)node * C);
    _Float16* drow = ef + r * LDP + h * C;
#pragma unroll
    for (int j = 0; j < 16; ++j) {
      float4 v = fp[j];
      drow[4 * j + 0] = (_Float16)v.x; drow[4 * j + 1] = (_Float16)v.y;
      drow[4 * j + 2] = (_Float16)v.z; drow[4 * j + 3] = (_Float16)v.w;
    }
  }
  __syncthreads();

  const int lane = tid & 31;
  const int wave = tid >> 5;
  const int row  = lane & 15;
  const int hi   = lane >> 4;

  v8f acc[8];
#pragma unroll
  for (int t = 0; t < 8; ++t) acc[t] = 0.f;

#pragma unroll
  for (int ks = 0; ks < 4; ++ks) {
    const int k0 = ks * 32;
    // A fragment (16x32 f16): lanes 0-15 K=kb..kb+7,16+kb..; kb = hi*8
    Frag a;
    const _Float16* ap = ef + (wave * 16 + row) * LDP + k0 + hi * 8;
    a.q[0] = *(const uint4*)ap;
    a.q[1] = *(const uint4*)(ap + 16);
#pragma unroll
    for (int t = 0; t < 8; ++t) {
      // B fragment (32x16): lane col n=t*16+row, K=kb..kb+7,16+kb..; kb = hi*8
      Frag b;
      const _Float16* bp = wlin + (t * 16 + row) * C2 + k0 + hi * 8;
      b.q[0] = *(const uint4*)bp;
      b.q[1] = *(const uint4*)(bp + 16);
      acc[t] = __builtin_amdgcn_wmma_f32_16x16x32_f16(
          false, a.v, false, b.v, (short)0, acc[t], false, false);
    }
  }

  // C layout: vgpr i, lanes 0-15 -> M=i; lanes 16-31 -> M=i+8; N=row+t*16
#pragma unroll
  for (int t = 0; t < 8; ++t) {
    const int ch = t * 16 + row;
    const float bias = b_lin[ch];
#pragma unroll
    for (int i = 0; i < 8; ++i) {
      const int el = wave * 16 + hi * 8 + i;
      float a = acc[t][i] + bias;
      a = a > 0.f ? a : 0.f;
      const int d = s_dst[el];
      if (PASS == 1) {
        atomicMax(nmax_u + (size_t)d * C2 + ch, __float_as_uint(a));
      } else {
        const float ex = __expf(a - nmax_f[(size_t)d * C2 + ch]);
        const int s = s_src[el];
        const float efv = (ch < C) ? features[(size_t)s * C + ch]
                                   : features[(size_t)d * C + (ch - C)];
        atomicAdd(den + (size_t)d * C2 + ch, ex);
        atomicAdd(num + (size_t)d * C2 + ch, ex * efv);
      }
    }
  }
}

// ---------------- K3: out = (num/den) @ Wout^T + b_out ----------------
__global__ __launch_bounds__(256) void out_kernel(
    const float* __restrict__ den, const float* __restrict__ num,
    const unsigned short* __restrict__ wout_u,
    const float* __restrict__ b_out, float* __restrict__ out) {
  __shared__ __align__(16) _Float16 nf[128 * LDP];
  const _Float16* wout = (const _Float16*)wout_u;
  const int tid = threadIdx.x;
  const int nbase = blockIdx.x * 128;

  { // new_features = den > 0 ? num/den : 0  -> LDS f16
    const int r = tid >> 1, h = tid & 1;
    const int node = nbase + r;
    _Float16* drow = nf + r * LDP + h * C;
    if (node < NNODES) {
      const float4* dp = (const float4*)(den + (size_t)node * C2 + h * C);
      const float4* np = (const float4*)(num + (size_t)node * C2 + h * C);
#pragma unroll
      for (int j = 0; j < 16; ++j) {
        float4 dv = dp[j], nv = np[j];
        drow[4 * j + 0] = (_Float16)(dv.x > 0.f ? nv.x / dv.x : 0.f);
        drow[4 * j + 1] = (_Float16)(dv.y > 0.f ? nv.y / dv.y : 0.f);
        drow[4 * j + 2] = (_Float16)(dv.z > 0.f ? nv.z / dv.z : 0.f);
        drow[4 * j + 3] = (_Float16)(dv.w > 0.f ? nv.w / dv.w : 0.f);
      }
    } else {
#pragma unroll
      for (int j = 0; j < 64; ++j) drow[j] = (_Float16)0.f;
    }
  }
  __syncthreads();

  const int lane = tid & 31;
  const int wave = tid >> 5;
  const int row  = lane & 15;
  const int hi   = lane >> 4;

  v8f acc[4];
#pragma unroll
  for (int t = 0; t < 4; ++t) acc[t] = 0.f;

#pragma unroll
  for (int ks = 0; ks < 4; ++ks) {
    const int k0 = ks * 32;
    Frag a;
    const _Float16* ap = nf + (wave * 16 + row) * LDP + k0 + hi * 8;
    a.q[0] = *(const uint4*)ap;
    a.q[1] = *(const uint4*)(ap + 16);
#pragma unroll
    for (int t = 0; t < 4; ++t) {
      Frag b;
      const _Float16* bp = wout + (t * 16 + row) * C2 + k0 + hi * 8;
      b.q[0] = *(const uint4*)bp;
      b.q[1] = *(const uint4*)(bp + 16);
      acc[t] = __builtin_amdgcn_wmma_f32_16x16x32_f16(
          false, a.v, false, b.v, (short)0, acc[t], false, false);
    }
  }

#pragma unroll
  for (int t = 0; t < 4; ++t) {
    const int ch = t * 16 + row;
    const float bias = b_out[ch];
#pragma unroll
    for (int i = 0; i < 8; ++i) {
      const int m = wave * 16 + hi * 8 + i;
      const int node = nbase + m;
      if (node < NNODES) out[(size_t)node * C + ch] = acc[t][i] + bias;
    }
  }
}

extern "C" void kernel_launch(void* const* d_in, const int* in_sizes, int n_in,
                              void* d_out, int out_size, void* d_ws, size_t ws_size,
                              hipStream_t stream) {
  const float* features = (const float*)d_in[0];
  const float* W_lin    = (const float*)d_in[1];
  const float* b_lin    = (const float*)d_in[2];
  const float* W_out    = (const float*)d_in[3];
  const float* b_out    = (const float*)d_in[4];
  const int*   src      = (const int*)d_in[5];
  const int*   dst      = (const int*)d_in[6];
  float* out = (float*)d_out;

  char* ws = (char*)d_ws;
  unsigned short* wlin_h = (unsigned short*)(ws + WLIN_OFF);
  unsigned short* wout_h = (unsigned short*)(ws + WOUT_OFF);
  unsigned* nmax_u = (unsigned*)(ws + NMAX_OFF);
  const float* nmax_f = (const float*)(ws + NMAX_OFF);
  float* den = (float*)(ws + DEN_OFF);
  float* num = (float*)(ws + NUM_OFF);

  const long zero_dwords = 3L * NNODES * C2;  // nmax + den + num (contiguous)
  init_kernel<<<2048, 256, 0, stream>>>(W_lin, W_out, wlin_h, wout_h,
                                        nmax_u, zero_dwords);

  const int eblocks = NEDGES / 128;  // 6250
  edge_kernel<1><<<eblocks, 256, 0, stream>>>(features, wlin_h, b_lin, src, dst,
                                              nmax_u, nmax_f, den, num);
  edge_kernel<2><<<eblocks, 256, 0, stream>>>(features, wlin_h, b_lin, src, dst,
                                              nmax_u, nmax_f, den, num);

  const int nblocks = (NNODES + 127) / 128;  // 391
  out_kernel<<<nblocks, 256, 0, stream>>>(den, num, wout_h, b_out, out);
}
